// MeanPoolAggregator_9182640078909
// MI455X (gfx1250) — matrix operand order repaired and checked by reading, not compiled, not run on hardware
//
#include <hip/hip_runtime.h>
#include <hip/hip_bf16.h>

// MeanPoolAggregator fused kernel for MI455X (gfx1250, wave32).
// out[n,p] = (1/16) * sum_j features[idx[n,j],:] . W[p,:]
//          = mean(features[idx[n,:],:]) @ W^T        (linearity of mean)
//
// Block = 128 threads (4 wave32 waves) handling 64 output nodes.
// Stage 0: convert W (128x256 f32) -> bf16 in LDS, row-major with padded rows (no swizzle).
// Stage 1: gather+mean -> bf16 tile in LDS (padded rows, bank-conflict free).
// Stage 2: per-wave 16x128 WMMA strip, v_wmma_f32_16x16x32_bf16 (f32 accum);
//          A and B fragments both come straight from LDS via ds_load_b128.

#define N_BATCH  50000
#define N_UNIQUE 100000
#define K_NEIGH  16
#define HID      256
#define POOL     128

#define NODES_PER_BLOCK 64   // 4 waves x 16-row WMMA strips
#define THREADS         128

// Padded row stride: 33 uint4 (528 B = 132 dwords; 132 % 64 = 4 -> lanes 0..15
// hit distinct LDS banks on ds_load_b128 fragment reads).
#define ROW_Q 33

typedef __attribute__((ext_vector_type(16))) __bf16 v16bf;
typedef __attribute__((ext_vector_type(8)))  float  v8f;

union AB16 { v16bf v; uint4 q[2]; };

__device__ __forceinline__ unsigned pack_bf16x2(float lo, float hi) {
    union { __bf16 h[2]; unsigned u; } p;
    p.h[0] = (__bf16)lo;   // clang emits v_cvt_pk_bf16_f32 (RNE)
    p.h[1] = (__bf16)hi;
    return p.u;
}

__global__ __launch_bounds__(THREADS)
void meanpool_wmma_kernel(const int*   __restrict__ neigh_idx,   // [N_BATCH, 16]
                          const float* __restrict__ features,    // [N_UNIQUE, 256]
                          const float* __restrict__ W,           // [128, 256]
                          float*       __restrict__ out)         // [N_BATCH, 128]
{
    __shared__ uint4 sA[NODES_PER_BLOCK * ROW_Q];   // 33,792 B: mean feats, bf16
    __shared__ uint4 sW[POOL * ROW_Q];              // 67,584 B: W, bf16, K-packed

    const int tid    = threadIdx.x;
    const int lane   = tid & 31;
    const int wave   = tid >> 5;          // 0..3, owns local rows [16w, 16w+16)
    const int colL   = lane & 15;         // column / row-in-strip selector
    const int hiHalf = lane >> 4;         // 0: lanes 0-15, 1: lanes 16-31
    const int n0     = blockIdx.x * NODES_PER_BLOCK;

    // ---------------- Stage 0: stage W -> bf16 LDS (once per block) ------------
    // 128 rows x 32 uint4 chunks = 4096 chunks; 128 threads x 32 iters,
    // consecutive tids read consecutive 32B of W -> fully coalesced.
    #pragma unroll 4
    for (int i = 0; i < 32; ++i) {
        const int f  = i * THREADS + tid;
        const int r  = f >> 5;            // W row (pool channel) 0..127
        const int ch = f & 31;            // 8-float chunk within row
        const float* src = W + (long)r * HID + ch * 8;
        float4 f0 = ((const float4*)src)[0];
        float4 f1 = ((const float4*)src)[1];
        uint4 q;
        q.x = pack_bf16x2(f0.x, f0.y);
        q.y = pack_bf16x2(f0.z, f0.w);
        q.z = pack_bf16x2(f1.x, f1.y);
        q.w = pack_bf16x2(f1.z, f1.w);
        sW[r * ROW_Q + ch] = q;
    }

    // ------------------- Stage 1: gather + mean -> bf16 LDS -------------------
    // Lane l owns hid columns [8l, 8l+8): two float4 loads per neighbor row give
    // a fully coalesced 1KB read per wave. features (102.4 MB) is L2-resident.
    for (int t = 0; t < 16; ++t) {
        const int local = wave * 16 + t;
        int n = n0 + local;
        if (n >= N_BATCH) n = N_BATCH - 1;          // clamped gather; store guarded
        const int* idx_row = neigh_idx + (long)n * K_NEIGH;

        float a0 = 0.f, a1 = 0.f, a2 = 0.f, a3 = 0.f;
        float a4 = 0.f, a5 = 0.f, a6 = 0.f, a7 = 0.f;

        for (int j = 0; j < K_NEIGH; ++j) {
            int row = idx_row[j];
            if (row < 0) row = 0;
            if (row >= N_UNIQUE) row = N_UNIQUE - 1;
            const float* src = features + (long)row * HID + lane * 8;
            if (j + 1 < K_NEIGH) {                  // hide next gather's latency
                int nrow = idx_row[j + 1];
                __builtin_prefetch(features + (long)nrow * HID + lane * 8, 0, 1);
            }
            float4 f0 = ((const float4*)src)[0];
            float4 f1 = ((const float4*)src)[1];
            a0 += f0.x; a1 += f0.y; a2 += f0.z; a3 += f0.w;
            a4 += f1.x; a5 += f1.y; a6 += f1.z; a7 += f1.w;
        }

        const float s = 1.0f / (float)K_NEIGH;
        uint4 q;
        q.x = pack_bf16x2(a0 * s, a1 * s);
        q.y = pack_bf16x2(a2 * s, a3 * s);
        q.z = pack_bf16x2(a4 * s, a5 * s);
        q.w = pack_bf16x2(a6 * s, a7 * s);
        sA[local * ROW_Q + lane] = q;               // lane l -> bytes [16l,16l+16)
    }

    __syncthreads();   // sW (cross-wave) and sA visible before WMMA stage

    // ------------------- Stage 2: WMMA 16x128 strip per wave -------------------
    v8f acc[8] = {};   // 8 N-tiles of 16x16 f32 accumulators

    #pragma unroll
    for (int k = 0; k < 8; ++k) {                   // K-step: K = 32k .. 32k+31
        // A fragment (16x32 bf16), ISA layout (CDNA5 ISA 7.12.2):
        //  lanes 0-15 : M=lane,    V0-3 = K 0..7,   V4-7 = K 16..23
        //  lanes 16-31: M=lane-16, V0-3 = K 8..15,  V4-7 = K 24..31
        AB16 a;
        const int qbase = (wave * 16 + colL) * ROW_Q + 4 * k + hiHalf;
        a.q[0] = sA[qbase];
        a.q[1] = sA[qbase + 2];

        #pragma unroll
        for (int nt = 0; nt < 8; ++nt) {            // N-tile: cols 16*nt .. +15
            // B fragment (32x16 bf16): lane holds column c = 16*nt + colL,
            // K = 32k + 8*hiHalf .. +7 and 32k + 16 + 8*hiHalf .. +7 of sW row c.
            AB16 b;
            const int bbase = (nt * 16 + colL) * ROW_Q + 4 * k + hiHalf;
            b.q[0] = sW[bbase];
            b.q[1] = sW[bbase + 2];

            acc[nt] = __builtin_amdgcn_wmma_f32_16x16x32_bf16(
                /*neg_a=*/false, a.v, /*neg_b=*/false, b.v,
                /*c_mod=*/(short)0, acc[nt],
                /*reuse_a=*/false, /*reuse_b=*/false);
        }
    }

    // D layout: VGPR r -> (lanes 0-15: M=r, N=lane), (lanes 16-31: M=r+8, N=lane-16)
    #pragma unroll
    for (int nt = 0; nt < 8; ++nt) {
        const int c = nt * 16 + colL;
        #pragma unroll
        for (int r = 0; r < 8; ++r) {
            const int m = wave * 16 + r + hiHalf * 8;
            const int n = n0 + m;
            if (n < N_BATCH) out[(long)n * POOL + c] = acc[nt][r];
        }
    }
}

extern "C" void kernel_launch(void* const* d_in, const int* in_sizes, int n_in,
                              void* d_out, int out_size, void* d_ws, size_t ws_size,
                              hipStream_t stream) {
    (void)in_sizes; (void)n_in; (void)out_size; (void)d_ws; (void)ws_size;
    const int*   neigh = (const int*)d_in[0];
    const float* feats = (const float*)d_in[1];
    const float* W     = (const float*)d_in[2];
    float*       out   = (float*)d_out;

    const int grid = (N_BATCH + NODES_PER_BLOCK - 1) / NODES_PER_BLOCK;  // 782
    hipLaunchKernelGGL(meanpool_wmma_kernel, dim3(grid), dim3(THREADS), 0, stream,
                       neigh, feats, W, out);
}
